// RelationUnit_50233937494176
// MI455X (gfx1250) — matrix operand order repaired and checked by reading, not compiled, not run on hardware
//
#include <hip/hip_runtime.h>

#define N_PROP 1024
#define D_APP  2048
#define D_K    128
#define D_G    128

typedef __attribute__((ext_vector_type(16))) __bf16 v16bf;
typedef __attribute__((ext_vector_type(8)))  float  v8f;

union BF16x16 {
  unsigned short h[16];
  uint4          q[2];
  v16bf          v;
};

// fp32 -> bf16, round to nearest even
__device__ __forceinline__ unsigned short f2bf(float x) {
  unsigned int u = __float_as_uint(x);
  u += 0x7FFFu + ((u >> 16) & 1u);
  return (unsigned short)(u >> 16);
}

// ---- WMMA fragment loaders (layouts per CDNA5 ISA 7.12.2, wave32) ----
// A 16x32 bf16: lane L -> row M = L&15; K runs {b..b+7} and {b+16..b+23}, b = (L>=16)?8:0
__device__ __forceinline__ v16bf load_a16(const unsigned short* A, int rs,
                                          int m0, int k0, int lane) {
  int hi = lane >> 4, l = lane & 15;
  const unsigned short* p = A + (size_t)(m0 + l) * rs + k0 + hi * 8;
  BF16x16 a;
  a.q[0] = *(const uint4*)p;          // K = b .. b+7
  a.q[1] = *(const uint4*)(p + 16);   // K = b+16 .. b+23
  return a.v;
}
// B 32x16 bf16: lane L -> col N = L&15; K runs {b..b+7} and {b+16..b+23}, b = (L>=16)?8:0
// BT[n][k] must be contiguous in k with row stride rs.
__device__ __forceinline__ v16bf load_b16(const unsigned short* BT, int rs,
                                          int n0, int k0, int lane) {
  int hi = lane >> 4, l = lane & 15;
  const unsigned short* p = BT + (size_t)(n0 + l) * rs + k0 + hi * 8;
  BF16x16 b;
  b.q[0] = *(const uint4*)p;          // K = b .. b+7
  b.q[1] = *(const uint4*)(p + 16);   // K = b+16 .. b+23
  return b.v;
}

__device__ __forceinline__ v8f wmma_bf16(v16bf a, v16bf b, v8f c) {
  return __builtin_amdgcn_wmma_f32_16x16x32_bf16(false, a, false, b,
                                                 (short)0, c, false, false);
}

// ---------------- Kernel 1: fp32 -> bf16 staging -----------------
__global__ void __launch_bounds__(256) prep_bf16(
    const float* __restrict__ app,
    const float* __restrict__ wq, const float* __restrict__ wk,
    const float* __restrict__ wv,
    unsigned short* __restrict__ app_bf,
    unsigned short* __restrict__ wqw_bf,
    unsigned short* __restrict__ wkw_bf,
    unsigned short* __restrict__ wvw_bf) {
  int tid = blockIdx.x * blockDim.x + threadIdx.x;
  int nth = gridDim.x * blockDim.x;
  const int n_app = N_PROP * D_APP;
  const int n_w   = D_K * D_APP;
  for (int i = tid; i < n_app; i += nth) app_bf[i] = f2bf(app[i]);
  for (int i = tid; i < n_w; i += nth) {
    wqw_bf[i] = f2bf(wq[i]);
    wkw_bf[i] = f2bf(wk[i]);
    wvw_bf[i] = f2bf(wv[i]);
  }
}

// ------- Kernel 2: Q/K/V projections, bf16 WMMA, f32 accum -------
// out[m][j] = sum_k app[m][k] * W[j][k] + b[j]
// sel 0: q (scaled by 1/sqrt(128)), sel 1: k, sel 2: v stored transposed (j-major)
__global__ void __launch_bounds__(256) qkv_gemm(
    const unsigned short* __restrict__ app_bf,
    const unsigned short* __restrict__ wqw_bf,
    const unsigned short* __restrict__ wkw_bf,
    const unsigned short* __restrict__ wvw_bf,
    const float* __restrict__ bq, const float* __restrict__ bk,
    const float* __restrict__ bv,
    unsigned short* __restrict__ q_bf,
    unsigned short* __restrict__ k_bf,
    unsigned short* __restrict__ vT_bf) {
  int sel  = blockIdx.y;
  int lane = threadIdx.x & 31;
  int wave = threadIdx.x >> 5;
  int m0   = blockIdx.x * 128 + wave * 16;
  const unsigned short* W = (sel == 0) ? wqw_bf : ((sel == 1) ? wkw_bf : wvw_bf);
  const float* bias       = (sel == 0) ? bq     : ((sel == 1) ? bk     : bv);

  v8f acc[8] = {};
  for (int k0 = 0; k0 < D_APP; k0 += 32) {
    v16bf a = load_a16(app_bf, D_APP, m0, k0, lane);
#pragma unroll
    for (int t = 0; t < 8; ++t) {
      v16bf b = load_b16(W, D_APP, t * 16, k0, lane);
      acc[t] = wmma_bf16(a, b, acc[t]);
    }
  }
  int hi = lane >> 4, l = lane & 15;
#pragma unroll
  for (int t = 0; t < 8; ++t) {
    int j = t * 16 + l;
    float bj = bias[j];
#pragma unroll
    for (int r = 0; r < 8; ++r) {
      int m = m0 + r + hi * 8;
      float v = acc[t][r] + bj;
      if (sel == 0)      q_bf[(size_t)m * D_K + j] = f2bf(v * 0.08838834764831845f);
      else if (sel == 1) k_bf[(size_t)m * D_K + j] = f2bf(v);
      else               vT_bf[(size_t)j * N_PROP + m] = f2bf(v);
    }
  }
}

// -------- Kernel 3: geometric bias, streams 512MB at HBM rate --------
// G[m][n] = relu(dot(pos[m][n][:], wg) + bg)
// 8 lanes per (m,n) pair -> every b128 load covers 4 full cachelines.
__global__ void __launch_bounds__(256) geo_bias(
    const float* __restrict__ pos, const float* __restrict__ wg,
    const float* __restrict__ wgb, float* __restrict__ G) {
  int m    = blockIdx.x;
  int lane = threadIdx.x & 31;
  int wave = threadIdx.x >> 5;
  int qp   = lane >> 3;   // pair within wave iteration (0..3)
  int s    = lane & 7;    // sublane within pair
  const float4* wg4 = (const float4*)wg;
  float4 wgv[4];
#pragma unroll
  for (int i = 0; i < 4; ++i) wgv[i] = wg4[i * 8 + s];
  float bg = wgb[0];

  for (int it = 0; it < 32; ++it) {
    int n = it * 32 + wave * 4 + qp;
    const float4* p = (const float4*)(pos + ((size_t)m * N_PROP + n) * D_G);
    float acc = 0.f;
#pragma unroll
    for (int i = 0; i < 4; ++i) {
      float4 x = p[i * 8 + s];
      acc += x.x * wgv[i].x + x.y * wgv[i].y + x.z * wgv[i].z + x.w * wgv[i].w;
    }
    acc += __shfl_xor(acc, 1);
    acc += __shfl_xor(acc, 2);
    acc += __shfl_xor(acc, 4);
    if (s == 0) {
      float v = acc + bg;
      G[(size_t)m * N_PROP + n] = v > 0.f ? v : 0.f;
    }
  }
}

// --- Kernel 4: fused scores + bias + row softmax + (P @ w_v), 16 rows/WG ---
__global__ void __launch_bounds__(256) attn_fused(
    const unsigned short* __restrict__ q_bf,
    const unsigned short* __restrict__ k_bf,
    const unsigned short* __restrict__ vT_bf,
    const float* __restrict__ G, float* __restrict__ out) {
  __shared__ float          S[16][N_PROP];   // 64 KB score strip
  __shared__ unsigned short P[16][N_PROP];   // 32 KB bf16 probabilities
  __shared__ float          red[16][16];
  __shared__ float          rowstat[16];

  int tid  = threadIdx.x;
  int lane = tid & 31, wave = tid >> 5;
  int hi   = lane >> 4, l = lane & 15;
  int m0   = blockIdx.x * 16;

  // Phase A: S = q_scaled @ k^T  (each wave owns a 128-column slab)
  {
    int c0 = wave * 128;
    v8f acc[8] = {};
#pragma unroll
    for (int ks = 0; ks < 4; ++ks) {
      int k0 = ks * 32;
      v16bf a = load_a16(q_bf, D_K, m0, k0, lane);
#pragma unroll
      for (int t = 0; t < 8; ++t) {
        v16bf b = load_b16(k_bf, D_K, c0 + t * 16, k0, lane);
        acc[t] = wmma_bf16(a, b, acc[t]);
      }
    }
#pragma unroll
    for (int t = 0; t < 8; ++t)
#pragma unroll
      for (int r = 0; r < 8; ++r)
        S[r + hi * 8][c0 + t * 16 + l] = acc[t][r];
  }
  __syncthreads();

  // Phase B: add geometric bias, then row softmax (16 threads per row)
  int rr  = tid >> 4;   // row 0..15
  int sub = tid & 15;   // 16 threads per row, 64 cols each
  {
    float lmax = -3.0e38f;
    const float* Grow = G + (size_t)(m0 + rr) * N_PROP;
    for (int c = sub * 64; c < sub * 64 + 64; ++c) {
      float v = S[rr][c] + Grow[c];
      S[rr][c] = v;
      lmax = fmaxf(lmax, v);
    }
    red[rr][sub] = lmax;
  }
  __syncthreads();
  if (sub == 0) {
    float mx = red[rr][0];
#pragma unroll
    for (int i = 1; i < 16; ++i) mx = fmaxf(mx, red[rr][i]);
    rowstat[rr] = mx;
  }
  __syncthreads();
  {
    float mx = rowstat[rr];
    float lsum = 0.f;
    for (int c = sub * 64; c < sub * 64 + 64; ++c) {
      float e = __expf(S[rr][c] - mx);
      S[rr][c] = e;
      lsum += e;
    }
    red[rr][sub] = lsum;
  }
  __syncthreads();
  if (sub == 0) {
    float sm = 0.f;
#pragma unroll
    for (int i = 0; i < 16; ++i) sm += red[rr][i];
    rowstat[rr] = 1.0f / sm;
  }
  __syncthreads();
  {
    float inv = rowstat[rr];
    for (int c = sub * 64; c < sub * 64 + 64; ++c)
      P[rr][c] = f2bf(S[rr][c] * inv);
  }
  __syncthreads();

  // Phase D: out[m][j] = sum_n P[m][n] * v[n][j]; wave w owns 16-col tile
  {
    int j0 = wave * 16;
    v8f o = {};
    const unsigned short* Pp = &P[0][0];
    for (int ks = 0; ks < 32; ++ks) {
      int k0 = ks * 32;
      v16bf a = load_a16(Pp, N_PROP, 0, k0, lane);   // from LDS
      v16bf b = load_b16(vT_bf, N_PROP, j0, k0, lane);
      o = wmma_bf16(a, b, o);
    }
#pragma unroll
    for (int r = 0; r < 8; ++r)
      out[(size_t)(m0 + r + hi * 8) * D_K + j0 + l] = o[r];
  }
}

extern "C" void kernel_launch(void* const* d_in, const int* in_sizes, int n_in,
                              void* d_out, int out_size, void* d_ws, size_t ws_size,
                              hipStream_t stream) {
  const float* app = (const float*)d_in[0];
  const float* pos = (const float*)d_in[1];
  const float* WQw = (const float*)d_in[2];
  const float* WQb = (const float*)d_in[3];
  const float* WKw = (const float*)d_in[4];
  const float* WKb = (const float*)d_in[5];
  const float* WVw = (const float*)d_in[6];
  const float* WVb = (const float*)d_in[7];
  const float* WGw = (const float*)d_in[8];
  const float* WGb = (const float*)d_in[9];
  float* out = (float*)d_out;

  char* ws = (char*)d_ws;
  size_t off = 0;
  unsigned short* app_bf = (unsigned short*)(ws + off); off += (size_t)N_PROP * D_APP * 2;  // 4 MB
  unsigned short* wqw_bf = (unsigned short*)(ws + off); off += (size_t)D_K * D_APP * 2;     // 512 KB
  unsigned short* wkw_bf = (unsigned short*)(ws + off); off += (size_t)D_K * D_APP * 2;
  unsigned short* wvw_bf = (unsigned short*)(ws + off); off += (size_t)D_K * D_APP * 2;
  unsigned short* q_bf   = (unsigned short*)(ws + off); off += (size_t)N_PROP * D_K * 2;    // 256 KB
  unsigned short* k_bf   = (unsigned short*)(ws + off); off += (size_t)N_PROP * D_K * 2;
  unsigned short* vT_bf  = (unsigned short*)(ws + off); off += (size_t)N_PROP * D_K * 2;
  float*          G      = (float*)(ws + off);          off += (size_t)N_PROP * N_PROP * 4; // 4 MB

  prep_bf16<<<512, 256, 0, stream>>>(app, WQw, WKw, WVw,
                                     app_bf, wqw_bf, wkw_bf, wvw_bf);
  qkv_gemm<<<dim3(8, 3), 256, 0, stream>>>(app_bf, wqw_bf, wkw_bf, wvw_bf,
                                           WQb, WKb, WVb, q_bf, k_bf, vT_bf);
  geo_bias<<<N_PROP, 256, 0, stream>>>(pos, WGw, WGb, G);
  attn_fused<<<N_PROP / 16, 256, 0, stream>>>(q_bf, k_bf, vT_bf, G, out);
}
